// CSWinAttention_34694745817855
// MI455X (gfx1250) — compile-verified, matrix operand order checked by reading, not run on hardware
//
#include <hip/hip_runtime.h>
#include <hip/hip_bf16.h>

typedef __attribute__((ext_vector_type(16))) _Float16 v16h;
typedef __attribute__((ext_vector_type(8)))  float    v8f;

// ---------------------------------------------------------------------------
// WMMA fragment loaders (wave32, 16-bit A/B layout per CDNA5 ISA 7.12.2):
//   lane l holds row (l&15); khi = (l>>4)*8; half-element e maps to
//   K = 16*(e>=8) + khi + (e&7).
// B fragments (32x16 K-major) load identically from the row-major *transpose*
// of B, i.e. B^T (N x K) — which for Q·K^T and P·V is exactly the row-major
// tile we keep in LDS.
// ---------------------------------------------------------------------------

__device__ static inline v16h frag_k16(const _Float16* base, int ld) {
  // 16x16 tile, zero-padded to K=32 (head_dim == 16)
  const int lane = threadIdx.x & 31;
  const int row  = lane & 15;
  const int khi  = (lane >> 4) << 3;
  v16h a;
#pragma unroll
  for (int e = 0; e < 16; ++e) {
    if (e < 8) a[e] = base[row * ld + khi + (e & 7)];
    else       a[e] = (_Float16)0.0f;
  }
  return a;
}

__device__ static inline v16h frag_k32_f16(const _Float16* base, int ld) {
  const int lane = threadIdx.x & 31;
  const int row  = lane & 15;
  const int khi  = (lane >> 4) << 3;
  v16h a;
#pragma unroll
  for (int e = 0; e < 16; ++e) {
    int k = ((e >> 3) << 4) + khi + (e & 7);
    a[e] = base[row * ld + k];
  }
  return a;
}

__device__ static inline v16h frag_k32_f32(const float* base, int ld) {
  const int lane = threadIdx.x & 31;
  const int row  = lane & 15;
  const int khi  = (lane >> 4) << 3;
  v16h a;
#pragma unroll
  for (int e = 0; e < 16; ++e) {
    int k = ((e >> 3) << 4) + khi + (e & 7);
    a[e] = (_Float16)base[row * ld + k];
  }
  return a;
}

// ---------------------------------------------------------------------------
// Kernel 1: qkv = x @ qkv_w^T + qkv_b, scattered into window-gathered layout
//   q/k : f16 [1024 window-heads][512 tokens][16]
//   v   : f32 same shape (kept f32 for LePE conv precision)
// window-head index: wh = (((dir*16 + b)*8 + win)*4 + local_head)
// ---------------------------------------------------------------------------
__global__ __launch_bounds__(256) void cswin_qkv_kernel(
    const float* __restrict__ x, const float* __restrict__ w,
    const float* __restrict__ bias,
    _Float16* __restrict__ qbuf, _Float16* __restrict__ kbuf,
    float* __restrict__ vbuf)
{
  __shared__ _Float16 As[64 * 128];
  __shared__ _Float16 Ws[64 * 128];
  const int tid  = threadIdx.x;
  const int lane = tid & 31;
  const int wave = tid >> 5;                 // 8 waves (wave32)
  const int mbase = blockIdx.x * 64;         // token tile
  const int nbase = blockIdx.y * 64;         // output-channel tile (of 384)

  __builtin_prefetch(x + (mbase + 63) * 128, 0, 0);

  for (int i = tid; i < 64 * 128; i += 256) {
    int r = i >> 7, k = i & 127;
    As[i] = (_Float16)x[(mbase + r) * 128 + k];
    Ws[i] = (_Float16)w[(nbase + r) * 128 + k];
  }
  __syncthreads();

  for (int t = wave * 2; t < wave * 2 + 2; ++t) {  // 16 tiles / 8 waves
    const int mt = t >> 2, nt = t & 3;
    v8f c = {};
#pragma unroll
    for (int kc = 0; kc < 4; ++kc) {               // K = 128
      v16h a = frag_k32_f16(As + (mt * 16) * 128 + kc * 32, 128);
      v16h b = frag_k32_f16(Ws + (nt * 16) * 128 + kc * 32, 128);
      c = __builtin_amdgcn_wmma_f32_16x16x32_f16(false, a, false, b,
                                                 (short)0, c, false, false);
    }
    const int col   = nbase + nt * 16 + (lane & 15);
    const int rhalf = (lane >> 4) << 3;
    const int sel   = col >> 7;                    // 0=q 1=k 2=v
    const int cc    = col & 127;
    const int head  = cc >> 4, hd = cc & 15;
    const int dir   = head >> 2, hloc = head & 3;
#pragma unroll
    for (int r = 0; r < 8; ++r) {
      const int m = mbase + mt * 16 + r + rhalf;
      const float val = c[r] + bias[col];
      const int b_ = m >> 12, n = m & 4095;
      const int y = n >> 6, xq = n & 63;
      int win, pos;
      if (dir == 0) { win = y >> 3;  pos = ((y & 7) << 6) | xq; }
      else          { win = xq >> 3; pos = (y << 3) | (xq & 7); }
      const int wh  = ((dir * 16 + b_) * 8 + win) * 4 + hloc;
      const int off = (wh * 512 + pos) * 16 + hd;
      if      (sel == 0) qbuf[off] = (_Float16)val;
      else if (sel == 1) kbuf[off] = (_Float16)val;
      else               vbuf[off] = val;
    }
  }
}

// ---------------------------------------------------------------------------
// Kernel 2: stripe attention for one (window-head, 64-row M-block).
//   - LePE (depthwise 3x3 + v) into Vt (d-major f16 in LDS)
//   - S = Q K^T via WMMA (K=16 zero-padded to 32)
//   - row softmax (scale 0.25) on f32 score block in LDS
//   - O = P @ V_le via WMMA, K=512 split across wave pairs, LDS combine
// ---------------------------------------------------------------------------
__global__ __launch_bounds__(256) void cswin_attn_kernel(
    const _Float16* __restrict__ qbuf, const _Float16* __restrict__ kbuf,
    const float* __restrict__ vbuf,
    const float* __restrict__ dwh_w, const float* __restrict__ dwh_b,
    const float* __restrict__ dwv_w, const float* __restrict__ dwv_b,
    float* __restrict__ obuf)
{
  extern __shared__ char smem[];
  float*    Ss   = (float*)smem;                     // 64 x 512 scores (128KB)
  float*    cscr = Ss + 64 * 512;                    // 4 x 256 combine
  float*    red  = cscr + 4 * 256;                   // 64 x 4 partials
  _Float16* Qs   = (_Float16*)(red + 64 * 4);        // 64 x 16
  _Float16* Ks   = Qs + 64 * 16;                     // 512 x 16
  _Float16* Vt   = Ks + 512 * 16;                    // 16 x VTLD (d-major)
  const int VTLD = 520;

  const int tid  = threadIdx.x;
  const int lane = tid & 31;
  const int wave = tid >> 5;
  const int mb   = blockIdx.x;        // 0..7  (64-row M block)
  const int wh   = blockIdx.y;        // 0..1023
  const int dir  = wh >> 9;
  const int mbase = mb * 64;

  const _Float16* qw = qbuf + (wh * 512 + mbase) * 16;
  const _Float16* kw = kbuf + (size_t)wh * 512 * 16;
  const float*    vw = vbuf + (size_t)wh * 512 * 16;

  __builtin_prefetch(vw + tid * 16, 0, 0);

  for (int i = tid; i < 64 * 16;  i += 256) Qs[i] = qw[i];
  for (int i = tid; i < 512 * 16; i += 256) Ks[i] = kw[i];

  // LePE: Vt[d][p] = v + conv3x3_dw(v) + bias, zero-padded within the window
  const float* cw = dir ? dwv_w : dwh_w;
  const float* cb = dir ? dwv_b : dwh_b;
  const int ws = dir ? 8 : 64;
  const int hs = 512 / ws;
  for (int i = tid; i < 16 * 512; i += 256) {
    const int d = i >> 9, p = i & 511;
    const int r = p / ws, c = p % ws;
    float acc = cb[d];
#pragma unroll
    for (int dy = -1; dy <= 1; ++dy)
#pragma unroll
      for (int dx = -1; dx <= 1; ++dx) {
        const int rr = r + dy, c2 = c + dx;
        if (rr >= 0 && rr < hs && c2 >= 0 && c2 < ws)
          acc += vw[(rr * ws + c2) * 16 + d] * cw[d * 9 + (dy + 1) * 3 + (dx + 1)];
      }
    acc += vw[p * 16 + d];
    Vt[d * VTLD + p] = (_Float16)acc;
  }
  __syncthreads();

  // --- S = Q K^T : 4x32 tiles, 16 per wave ---
  for (int i = 0; i < 16; ++i) {
    const int t  = wave * 16 + i;
    const int mt = t >> 5, nt = t & 31;
    v16h a = frag_k16(Qs + mt * 16 * 16, 16);
    v16h b = frag_k16(Ks + nt * 16 * 16, 16);
    v8f c = {};
    c = __builtin_amdgcn_wmma_f32_16x16x32_f16(false, a, false, b,
                                               (short)0, c, false, false);
    const int rhalf = (lane >> 4) << 3;
    const int col   = nt * 16 + (lane & 15);
#pragma unroll
    for (int r = 0; r < 8; ++r)
      Ss[(mt * 16 + r + rhalf) * 512 + col] = c[r];
  }
  __syncthreads();

  // --- row softmax, scale = 16^-0.5 = 0.25, 4 threads per row ---
  {
    const float scale = 0.25f;
    const int row = tid >> 2, part = tid & 3;
    float* rp = Ss + row * 512 + part * 128;
    float pmax = -3.4e38f;
    for (int j = 0; j < 128; ++j) pmax = fmaxf(pmax, rp[j]);
    red[row * 4 + part] = pmax;
    __syncthreads();
    const float rmax = fmaxf(fmaxf(red[row * 4 + 0], red[row * 4 + 1]),
                             fmaxf(red[row * 4 + 2], red[row * 4 + 3]));
    __syncthreads();
    float psum = 0.0f;
    for (int j = 0; j < 128; ++j) {
      const float e = __expf(scale * (rp[j] - rmax));
      rp[j] = e;
      psum += e;
    }
    red[row * 4 + part] = psum;
    __syncthreads();
    const float inv = 1.0f / (red[row * 4 + 0] + red[row * 4 + 1] +
                              red[row * 4 + 2] + red[row * 4 + 3]);
    for (int j = 0; j < 128; ++j) rp[j] *= inv;
  }
  __syncthreads();

  // --- O = P @ V_le : 4 M-tiles, K=512 split across wave pairs ---
  const int mt    = wave & 3;
  const int khalf = wave >> 2;
  v8f c = {};
  for (int kc = khalf * 8; kc < khalf * 8 + 8; ++kc) {
    v16h a = frag_k32_f32(Ss + (mt * 16) * 512 + kc * 32, 512);
    v16h b = frag_k32_f16(Vt + kc * 32, VTLD);
    c = __builtin_amdgcn_wmma_f32_16x16x32_f16(false, a, false, b,
                                               (short)0, c, false, false);
  }
  if (wave >= 4) {
#pragma unroll
    for (int r = 0; r < 8; ++r) cscr[mt * 256 + r * 32 + lane] = c[r];
  }
  __syncthreads();
  if (wave < 4) {
    const int hloc = wh & 3, win = (wh >> 2) & 7, b_ = (wh >> 5) & 15;
    const int d = lane & 15;
    const int rhalf = (lane >> 4) << 3;
    const int ch = (dir * 4 + hloc) * 16 + d;   // concat: h-heads 0..63, v 64..127
#pragma unroll
    for (int r = 0; r < 8; ++r) {
      const float val = c[r] + cscr[mt * 256 + r * 32 + lane];
      const int pos = mbase + mt * 16 + r + rhalf;
      const int n = dir ? ((pos >> 3) * 64 + win * 8 + (pos & 7))
                        : (win * 512 + pos);
      obuf[(b_ * 4096 + n) * 128 + ch] = val;
    }
  }
}

// ---------------------------------------------------------------------------
// Kernel 3: out = attn_out @ proj_w^T + proj_b   (65536x128 @ 128x128)
// ---------------------------------------------------------------------------
__global__ __launch_bounds__(256) void cswin_proj_kernel(
    const float* __restrict__ a_in, const float* __restrict__ w,
    const float* __restrict__ bias, float* __restrict__ out)
{
  __shared__ _Float16 As[64 * 128];
  __shared__ _Float16 Ws[64 * 128];
  const int tid  = threadIdx.x;
  const int lane = tid & 31;
  const int wave = tid >> 5;
  const int mbase = blockIdx.x * 64;
  const int nbase = blockIdx.y * 64;

  for (int i = tid; i < 64 * 128; i += 256) {
    int r = i >> 7, k = i & 127;
    As[i] = (_Float16)a_in[(mbase + r) * 128 + k];
    Ws[i] = (_Float16)w[(nbase + r) * 128 + k];
  }
  __syncthreads();

  for (int t = wave * 2; t < wave * 2 + 2; ++t) {
    const int mt = t >> 2, nt = t & 3;
    v8f c = {};
#pragma unroll
    for (int kc = 0; kc < 4; ++kc) {
      v16h a = frag_k32_f16(As + (mt * 16) * 128 + kc * 32, 128);
      v16h b = frag_k32_f16(Ws + (nt * 16) * 128 + kc * 32, 128);
      c = __builtin_amdgcn_wmma_f32_16x16x32_f16(false, a, false, b,
                                                 (short)0, c, false, false);
    }
    const int col   = nbase + nt * 16 + (lane & 15);
    const int rhalf = (lane >> 4) << 3;
#pragma unroll
    for (int r = 0; r < 8; ++r) {
      const int m = mbase + mt * 16 + r + rhalf;
      out[m * 128 + col] = c[r] + bias[col];
    }
  }
}

// ---------------------------------------------------------------------------
extern "C" void kernel_launch(void* const* d_in, const int* in_sizes, int n_in,
                              void* d_out, int out_size, void* d_ws, size_t ws_size,
                              hipStream_t stream) {
  const float* x      = (const float*)d_in[0];
  const float* qkv_w  = (const float*)d_in[1];
  const float* qkv_b  = (const float*)d_in[2];
  const float* proj_w = (const float*)d_in[3];
  const float* proj_b = (const float*)d_in[4];
  const float* dwh_w  = (const float*)d_in[5];
  const float* dwh_b  = (const float*)d_in[6];
  const float* dwv_w  = (const float*)d_in[7];
  const float* dwv_b  = (const float*)d_in[8];
  // d_in[9]=H, d_in[10]=W fixed at 64 for this problem size.

  char* wsb = (char*)d_ws;
  _Float16* qbuf = (_Float16*)(wsb);                       // 16 MB
  _Float16* kbuf = (_Float16*)(wsb + (size_t)16 * 1024 * 1024);  // 16 MB
  float*    vbuf = (float*)   (wsb + (size_t)32 * 1024 * 1024);  // 32 MB
  float*    obuf = (float*)   (wsb + (size_t)64 * 1024 * 1024);  // 32 MB

  // QKV projection + window-gather scatter
  cswin_qkv_kernel<<<dim3(1024, 6), 256, 0, stream>>>(
      x, qkv_w, qkv_b, qbuf, kbuf, vbuf);

  // Stripe attention: 1024 window-heads x 8 M-blocks
  const size_t smem = (size_t)64 * 512 * 4 + 4 * 256 * 4 + 64 * 4 * 4 +
                      (size_t)(64 * 16 + 512 * 16 + 16 * 520) * 2;
  cswin_attn_kernel<<<dim3(8, 1024), 256, smem, stream>>>(
      qbuf, kbuf, vbuf, dwh_w, dwh_b, dwv_w, dwv_b, obuf);

  // Output projection
  cswin_proj_kernel<<<dim3(1024, 2), 256, 0, stream>>>(
      obuf, proj_w, proj_b, (float*)d_out);
}
